// NGCFLayer_32229434589221
// MI455X (gfx1250) — matrix operand order repaired and checked by reading, not compiled, not run on hardware
//
#include <hip/hip_runtime.h>

#define D 128

typedef __attribute__((ext_vector_type(16))) _Float16 v16h;
typedef __attribute__((ext_vector_type(8)))  float    v8f;
typedef __attribute__((ext_vector_type(4)))  float    f4;

// ---------------- degree / normalization ----------------

__global__ void zero_f32(float* __restrict__ p, int n) {
  int i = blockIdx.x * blockDim.x + threadIdx.x;
  if (i < n) p[i] = 0.0f;
}

__global__ void deg_accum(const int* __restrict__ row, const int* __restrict__ col,
                          const float* __restrict__ vals,
                          float* __restrict__ rs, float* __restrict__ cs, int E) {
  int e = blockIdx.x * blockDim.x + threadIdx.x;
  if (e < E) {
    float v = vals[e];
    __hip_atomic_fetch_add(&rs[row[e]], v, __ATOMIC_RELAXED, __HIP_MEMORY_SCOPE_AGENT);
    __hip_atomic_fetch_add(&cs[col[e]], v, __ATOMIC_RELAXED, __HIP_MEMORY_SCOPE_AGENT);
  }
}

__global__ void deg_finish(float* __restrict__ rs, float* __restrict__ cs, int N) {
  int i = blockIdx.x * blockDim.x + threadIdx.x;
  if (i < N) {
    float r = rs[i]; rs[i] = rsqrtf(r > 0.0f ? r : 1.0f);
    float c = cs[i]; cs[i] = rsqrtf(c > 0.0f ? c : 1.0f);
  }
}

// ---------------- weight-fragment pre-pack (done once; 16 KB, L2/L0 resident) --
// B fragment layout for V_WMMA_F32_16X16X32_F16:
//   lane owns column n = lane&15; element v holds K = chunk*32 + 8*(lane>>4) + (v&7) + 16*(v>>3).
// Store, for each (matrix, hi/lo part, chunk, lane), the 16 f16 values contiguously
// so the GEMM loop fetches a whole fragment as one 32B vector load.
// Index: (((mat*2 + part)*4 + chunk)*32 + lane)*16 + elem

__global__ void pack_wfrag(const float* __restrict__ w1, const float* __restrict__ w2,
                           _Float16* __restrict__ bf) {
  int t = blockIdx.x * blockDim.x + threadIdx.x;   // 2*4*32*16 = 4096 items
  if (t >= 4096) return;
  int elem = t & 15;
  int lane = (t >> 4) & 31;
  int ci   = (t >> 9) & 3;
  int mat  = (t >> 11) & 1;
  int hs   = lane >> 4;
  int ncol = lane & 15;
  int k = ci * 32 + 8 * hs + (elem & 7) + ((elem >> 3) << 4);
  float v = (mat ? w2 : w1)[(size_t)k * D + ncol];
  _Float16 h = (_Float16)v;
  _Float16 l = (_Float16)(v - (float)h);
  bf[(size_t)((((mat * 2 + 0) * 4 + ci) * 32 + lane) * 16) + elem] = h;
  bf[(size_t)((((mat * 2 + 1) * 4 + ci) * 32 + lane) * 16) + elem] = l;
}

// ---------------- dense GEMMs via WMMA (f16 hi/lo split for fp32 accuracy) ----
// y1 = x @ w1            -> written to out (accumulator for scatter pass)
// S  = y1 + (x*x) @ w2   -> scratch; spmm(y1)+spmm(y2) == spmm(S)

__global__ void __launch_bounds__(256)
ngcf_gemm_wmma(const float* __restrict__ x, const v16h* __restrict__ BF,
               float* __restrict__ out, float* __restrict__ S, int N) {
  const int lane = threadIdx.x & 31;
  const int wave = threadIdx.x >> 5;
  const int tile = blockIdx.x * (blockDim.x >> 5) + wave;
  if (tile >= ((N + 15) >> 4)) return;           // whole-wave uniform: EXEC stays all-1s
  const int m0   = tile << 4;
  const int hs   = lane >> 4;                    // half-wave select
  const int mrow = lane & 15;                    // A row owned by this lane
  const int ncol = lane & 15;                    // B/C column owned by this lane

  v8f c1 = {};                                   // accum for x@w1
  v8f c2 = {};                                   // accum for (x*x)@w2
  const float* arow = x + (size_t)(m0 + mrow) * D;

  #pragma unroll
  for (int ci = 0; ci < 4; ++ci) {
    const int kk = ci * 32;
    v16h a1h, a1l, a2h, a2l;

    // A fragment (documented layout): elems 0..7 -> K = kk + 8*hs + j,
    //                                 elems 8..15 -> K = kk + 16 + 8*hs + j.
    // Each 8-run is 32B contiguous & 32B aligned -> explicit float4 loads.
    {
      const f4* p0 = (const f4*)(arow + kk + 8 * hs);        // K run 0..7
      const f4* p1 = (const f4*)(arow + kk + 16 + 8 * hs);   // K run 16..23 (+8*hs)
      f4 r0 = p0[0], r1 = p0[1], r2 = p1[0], r3 = p1[1];
      float av[16] = { r0.x, r0.y, r0.z, r0.w, r1.x, r1.y, r1.z, r1.w,
                       r2.x, r2.y, r2.z, r2.w, r3.x, r3.y, r3.z, r3.w };
      #pragma unroll
      for (int j = 0; j < 16; ++j) {
        float a = av[j];
        _Float16 h = (_Float16)a;
        a1h[j] = h; a1l[j] = (_Float16)(a - (float)h);
        float q = a * a;
        _Float16 hq = (_Float16)q;
        a2h[j] = hq; a2l[j] = (_Float16)(q - (float)hq);
      }
    }

    // B fragments: pre-packed, one 32B vector load each.
    v16h b1h = BF[((0 * 4 + ci) * 32) + lane];
    v16h b1l = BF[((1 * 4 + ci) * 32) + lane];
    v16h b2h = BF[((2 * 4 + ci) * 32) + lane];
    v16h b2l = BF[((3 * 4 + ci) * 32) + lane];

    // A@B ~= Ah@Bh + Ah@Bl + Al@Bh  (drop lo*lo; ~21-bit effective mantissa)
    c1 = __builtin_amdgcn_wmma_f32_16x16x32_f16(false, a1h, false, b1h, (short)0, c1, false, false);
    c1 = __builtin_amdgcn_wmma_f32_16x16x32_f16(false, a1h, false, b1l, (short)0, c1, false, false);
    c1 = __builtin_amdgcn_wmma_f32_16x16x32_f16(false, a1l, false, b1h, (short)0, c1, false, false);
    c2 = __builtin_amdgcn_wmma_f32_16x16x32_f16(false, a2h, false, b2h, (short)0, c2, false, false);
    c2 = __builtin_amdgcn_wmma_f32_16x16x32_f16(false, a2h, false, b2l, (short)0, c2, false, false);
    c2 = __builtin_amdgcn_wmma_f32_16x16x32_f16(false, a2l, false, b2h, (short)0, c2, false, false);
  }

  // C/D layout: VGPR r -> M = r + 8*hs, N = lane&15
  #pragma unroll
  for (int r = 0; r < 8; ++r) {
    size_t o = (size_t)(m0 + r + 8 * hs) * D + ncol;
    float y1 = c1[r];
    out[o] = y1;            // accumulator base = h_self pre-bias
    S[o]   = y1 + c2[r];    // combined feature to scatter once
  }
}

// ---------------- edge scatter: out[row] += nv * S[col] ----------------
// One wave per edge; lane handles 4 consecutive dims (512B gather / wave).
// S (51 MB) is resident in the 192 MB L2, so gathers and atomics stay on-chip.

__global__ void __launch_bounds__(256)
scatter_spmm(const int* __restrict__ row, const int* __restrict__ col,
             const float* __restrict__ vals, const float* __restrict__ rs,
             const float* __restrict__ cs, const float* __restrict__ S,
             float* __restrict__ out, int E) {
  int gid  = blockIdx.x * blockDim.x + threadIdx.x;
  int e    = gid >> 5;
  int lane = threadIdx.x & 31;
  if (e >= E) return;
  int r = row[e], c = col[e];
  float w = vals[e] * rs[r] * cs[c];             // same-address loads broadcast in HW
  f4 s = ((const f4*)(S + (size_t)c * D))[lane];
  float* dst = out + (size_t)r * D + lane * 4;
  __hip_atomic_fetch_add(dst + 0, w * s.x, __ATOMIC_RELAXED, __HIP_MEMORY_SCOPE_AGENT);
  __hip_atomic_fetch_add(dst + 1, w * s.y, __ATOMIC_RELAXED, __HIP_MEMORY_SCOPE_AGENT);
  __hip_atomic_fetch_add(dst + 2, w * s.z, __ATOMIC_RELAXED, __HIP_MEMORY_SCOPE_AGENT);
  __hip_atomic_fetch_add(dst + 3, w * s.w, __ATOMIC_RELAXED, __HIP_MEMORY_SCOPE_AGENT);
}

// ---------------- bias + leaky_relu epilogue ----------------

__global__ void bias_leaky(float* __restrict__ out, const float* __restrict__ b1,
                           const float* __restrict__ b2, int total) {
  int i = blockIdx.x * blockDim.x + threadIdx.x;
  if (i < total) {
    int d = i & (D - 1);
    float v = out[i] + 2.0f * b1[d] + b2[d];     // b1 appears in h_self and h_neigh
    out[i] = v > 0.0f ? v : 0.2f * v;
  }
}

// ---------------- launch ----------------

extern "C" void kernel_launch(void* const* d_in, const int* in_sizes, int n_in,
                              void* d_out, int out_size, void* d_ws, size_t ws_size,
                              hipStream_t stream) {
  const int*   row  = (const int*)  d_in[0];
  const int*   col  = (const int*)  d_in[1];
  const float* vals = (const float*)d_in[2];
  const float* x    = (const float*)d_in[3];
  const float* w1   = (const float*)d_in[4];
  const float* b1   = (const float*)d_in[5];
  const float* w2   = (const float*)d_in[6];
  const float* b2   = (const float*)d_in[7];
  float* out = (float*)d_out;

  const int E = in_sizes[0];
  const int N = in_sizes[3] / D;

  float* rs = (float*)d_ws;     // N floats: rsqrt(row degree)
  float* cs = rs + N;           // N floats: rsqrt(col degree)
  float* S  = cs + N;           // N*D floats: y1 + y2 (L2-resident gather source)
  // 64B-aligned weight-fragment table after S (16 KB)
  uintptr_t bfp = (uintptr_t)(S + (size_t)N * D);
  bfp = (bfp + 63) & ~(uintptr_t)63;
  _Float16* bfrag = (_Float16*)bfp;

  zero_f32  <<<(2 * N + 255) / 256, 256, 0, stream>>>(rs, 2 * N);
  deg_accum <<<(E + 255) / 256,     256, 0, stream>>>(row, col, vals, rs, cs, E);
  deg_finish<<<(N + 255) / 256,     256, 0, stream>>>(rs, cs, N);

  pack_wfrag<<<16, 256, 0, stream>>>(w1, w2, bfrag);

  const int tiles = (N + 15) / 16;
  ngcf_gemm_wmma<<<(tiles + 7) / 8, 256, 0, stream>>>(x, (const v16h*)bfrag, out, S, N);

  const long long sthreads = (long long)E * 32;
  scatter_spmm<<<(int)((sthreads + 255) / 256), 256, 0, stream>>>(row, col, vals, rs, cs, S, out, E);

  const int total = N * D;
  bias_leaky<<<(total + 255) / 256, 256, 0, stream>>>(out, b1, b2, total);
}
